// GRUseq2seq_28174985462639
// MI455X (gfx1250) — matrix operand order repaired and checked by reading, not compiled, not run on hardware
//
#include <hip/hip_runtime.h>
#include <hip/hip_bf16.h>
#include <math.h>

// ---------------------------------------------------------------------------
// GRU seq2seq for MI455X (gfx1250): all matrix math through
// v_wmma_f32_16x16x32_bf16 (bf16 in, f32 accumulate), register-double-buffered
// fragment streams so VMEM latency hides under the WMMA blocks.
// B=16, S=T=256, E=1024, H=512, ZHV=32000.
// ---------------------------------------------------------------------------

typedef __bf16 bf16_t;
typedef __attribute__((ext_vector_type(16))) __bf16 v16bf;
typedef __attribute__((ext_vector_type(8)))  __bf16 v8bf;
typedef __attribute__((ext_vector_type(8)))  float  v8f;

#define WMMA_BF16(a, b, c) \
  __builtin_amdgcn_wmma_f32_16x16x32_bf16(false, (a), false, (b), (short)0, (c), false, false)

// ISA 7.12.2 fragment layouts (wave32):
//  A (16x32, row-major [M,K]):   lane L -> row M=L&15; a[i] holds
//      K = kk + i + 8*((i>=8) + (L>=16))  => two contiguous 16B runs at
//      (kk + 8*(L>>4)) and (+16).
//  B (32x16 from W stored [N,K]): lane L -> col N=L&15; b[i] holds
//      K = kk + i + 8*((i>=8) + (L>=16))  => two contiguous 16B runs (same as A).
//  C/D (16x16 f32): c[v] = C[M = v + 8*(L>=16)][N = L&15].

__device__ __forceinline__ v16bf frag_a_rowmajor(const bf16_t* p /*rowptr + kk + 8*ksel*/) {
  v8bf lo = *reinterpret_cast<const v8bf*>(p);
  v8bf hi = *reinterpret_cast<const v8bf*>(p + 16);
  v16bf a;
#pragma unroll
  for (int i = 0; i < 8; ++i) { a[i] = lo[i]; a[i + 8] = hi[i]; }
  return a;
}

__device__ __forceinline__ v16bf frag_b_nk(const bf16_t* W, size_t ldb, int n0, int kk, int lane) {
  const bf16_t* p = W + (size_t)(n0 + (lane & 15)) * ldb + kk + 8 * (lane >> 4);
  v8bf lo = *reinterpret_cast<const v8bf*>(p);
  v8bf hi = *reinterpret_cast<const v8bf*>(p + 16);
  v16bf b;
#pragma unroll
  for (int i = 0; i < 8; ++i) { b[i] = lo[i]; b[i + 8] = hi[i]; }
  return b;
}

__device__ __forceinline__ float sigmoidf_(float x) { return 1.f / (1.f + __expf(-x)); }

// ---------------------------------------------------------------------------
// Elementwise f32 -> bf16 cast (weights; L2-resident thereafter)
// ---------------------------------------------------------------------------
__global__ void k_cast_bf16(const float* __restrict__ s, bf16_t* __restrict__ d, size_t n) {
  size_t i = (size_t)blockIdx.x * blockDim.x + threadIdx.x;
  if (i < n) d[i] = (bf16_t)s[i];
}

// ---------------------------------------------------------------------------
// Embedding gathers -> bf16 activations
// ---------------------------------------------------------------------------
__global__ void k_embed_enc(const int* __restrict__ sen, const float* __restrict__ emb,
                            bf16_t* __restrict__ out, int rows, int width) {
  size_t i = (size_t)blockIdx.x * blockDim.x + threadIdx.x;
  if (i >= (size_t)rows * width) return;
  int r = (int)(i / width), c = (int)(i % width);
  out[i] = (bf16_t)emb[(size_t)sen[r] * width + c];
}

__global__ void k_embed_dec(const int* __restrict__ zh, const float* __restrict__ emb,
                            bf16_t* __restrict__ out) {
  // rows = 16*256, width = 1024; shift-right with SOS = ZHV-2
  size_t i = (size_t)blockIdx.x * blockDim.x + threadIdx.x;
  if (i >= (size_t)16 * 256 * 1024) return;
  int r = (int)(i >> 10), c = (int)(i & 1023);
  int b = r >> 8, t = r & 255;
  int tok = (t == 0) ? (32000 - 2) : zh[b * 256 + t - 1];
  out[i] = (bf16_t)emb[(size_t)tok * 1024 + c];
}

// ---------------------------------------------------------------------------
// Generic GEMM: C[M,N] (f32) = A[M,K](bf16) @ W[N,K]^T(bf16) + bias[N]
// Block = 256 thr (8 waves, 4 wm x 2 wn). Wave tile 32x64 (2 A-frags x
// 4 B-frags -> 8 WMMA per chunk), block tile 128x128. K-loop ping-pongs two
// fragment register sets over 64-wide chunks so loads for chunk kk+32 are in
// flight while chunk kk feeds the matrix pipe. M % 128 == 0, N % 128 == 0,
// K % 64 == 0 for every call here.
// ---------------------------------------------------------------------------
__global__ __launch_bounds__(256) void k_gemm_bf16(
    const bf16_t* __restrict__ A, const bf16_t* __restrict__ W,
    const float* __restrict__ bias, float* __restrict__ C, int M, int N, int K) {
  const int lane = threadIdx.x & 31;
  const int wave = threadIdx.x >> 5;
  const int wm = wave & 3, wn = wave >> 2;
  const int m0 = blockIdx.y * 128 + wm * 32;
  const int n0 = blockIdx.x * 128 + wn * 64;
  if (m0 >= M || n0 >= N) return;
  const int lrow = lane & 15, ksel = lane >> 4;

  v8f acc[2][4];
#pragma unroll
  for (int i = 0; i < 2; ++i)
#pragma unroll
    for (int j = 0; j < 4; ++j) acc[i][j] = v8f{};

  const bf16_t* ar[2];
  ar[0] = A + (size_t)(m0 + lrow) * K + 8 * ksel;
  ar[1] = ar[0] + (size_t)16 * K;

  v16bf a0[2], b0[4], a1[2], b1[4];
#pragma unroll
  for (int i = 0; i < 2; ++i) a0[i] = frag_a_rowmajor(ar[i]);
#pragma unroll
  for (int j = 0; j < 4; ++j) b0[j] = frag_b_nk(W, (size_t)K, n0 + 16 * j, 0, lane);

  for (int kk = 0; kk < K; kk += 64) {
    // stage chunk kk+32 while computing chunk kk
#pragma unroll
    for (int i = 0; i < 2; ++i) a1[i] = frag_a_rowmajor(ar[i] + kk + 32);
#pragma unroll
    for (int j = 0; j < 4; ++j) b1[j] = frag_b_nk(W, (size_t)K, n0 + 16 * j, kk + 32, lane);
#pragma unroll
    for (int i = 0; i < 2; ++i)
#pragma unroll
      for (int j = 0; j < 4; ++j) acc[i][j] = WMMA_BF16(a0[i], b0[j], acc[i][j]);
    if (kk + 64 < K) {
#pragma unroll
      for (int i = 0; i < 2; ++i) a0[i] = frag_a_rowmajor(ar[i] + kk + 64);
#pragma unroll
      for (int j = 0; j < 4; ++j) b0[j] = frag_b_nk(W, (size_t)K, n0 + 16 * j, kk + 64, lane);
    }
#pragma unroll
    for (int i = 0; i < 2; ++i)
#pragma unroll
      for (int j = 0; j < 4; ++j) acc[i][j] = WMMA_BF16(a1[i], b1[j], acc[i][j]);
  }

#pragma unroll
  for (int j = 0; j < 4; ++j) {
    const int n = n0 + 16 * j + lrow;
    const float bn = bias ? bias[n] : 0.f;
#pragma unroll
    for (int i = 0; i < 2; ++i)
#pragma unroll
      for (int v = 0; v < 8; ++v)
        C[(size_t)(m0 + 16 * i + v + 8 * ksel) * N + n] = acc[i][j][v] + bn;
  }
}

// ---------------------------------------------------------------------------
// GRU recurrent scan, one layer. Single 1024-thread workgroup (32 waves),
// grid(1). Wave w owns hidden slice j0 = 16*w and keeps its r/z/n WMMA
// accumulators (N-tiles j0, j0+512, j0+1024) in registers -> gates computed
// in-register, no hp staging. h kept in LDS twice: f32 state + padded bf16
// WMMA operand (row stride 520 elems = 1040B -> 16 distinct bank quads).
// Whh fragment stream is register double-buffered (this is the serial
// latency chain of the whole net).
//   hp = h @ Whh^T + bhh ; r=sig(xr+hr) z=sig(xz+hz) n=tanh(xn+r*hn)
//   h' = (1-z)*n + z*h ; xp already contains x@Wih^T + bih.
// ---------------------------------------------------------------------------
#define HB_PAD 520
#define HF_PAD 516

__global__ __launch_bounds__(1024) void k_gru_scan(
    const float* __restrict__ xp,     // [B*T, 1536] f32 (bih already added)
    const bf16_t* __restrict__ Whh,   // [1536, 512] bf16
    const float* __restrict__ bhh,    // [1536]
    const float* __restrict__ h_init, // [16, 512] f32
    bf16_t* __restrict__ ys,          // [B*T, ldy] bf16 (cols 0..511 written)
    int ldy,
    float* __restrict__ h_out,        // [16,512] f32 or nullptr
    int T) {
  __shared__ __align__(16) bf16_t h_b[16][HB_PAD];
  __shared__ float h_f[16][HF_PAD];

  const int tid = threadIdx.x;
  const int lane = tid & 31, wave = tid >> 5;
  const int lrow = lane & 15, ksel = lane >> 4;
  const int j0 = wave * 16;
  const int j = j0 + lrow;

  for (int i = tid; i < 16 * 512; i += 1024) {
    int b = i >> 9, c = i & 511;
    float h = h_init[i];
    h_f[b][c] = h;
    h_b[b][c] = (bf16_t)h;
  }
  __syncthreads();

  const float bhr = bhh[j], bhz = bhh[j + 512], bhn = bhh[j + 1024];
  const bf16_t* hrow = &h_b[lrow][8 * ksel];

  for (int t = 0; t < T; ++t) {
    v8f hr = {}, hz = {}, hn = {};
    // ping-pong over 64-wide K chunks; Whh frags prefetched one chunk ahead
    v16bf a0 = frag_a_rowmajor(hrow);
    v16bf r0 = frag_b_nk(Whh, 512, j0, 0, lane);
    v16bf z0 = frag_b_nk(Whh, 512, j0 + 512, 0, lane);
    v16bf g0 = frag_b_nk(Whh, 512, j0 + 1024, 0, lane);
    for (int kk = 0; kk < 512; kk += 64) {
      v16bf a1 = frag_a_rowmajor(hrow + kk + 32);
      v16bf r1 = frag_b_nk(Whh, 512, j0, kk + 32, lane);
      v16bf z1 = frag_b_nk(Whh, 512, j0 + 512, kk + 32, lane);
      v16bf g1 = frag_b_nk(Whh, 512, j0 + 1024, kk + 32, lane);
      hr = WMMA_BF16(a0, r0, hr);
      hz = WMMA_BF16(a0, z0, hz);
      hn = WMMA_BF16(a0, g0, hn);
      if (kk + 64 < 512) {
        a0 = frag_a_rowmajor(hrow + kk + 64);
        r0 = frag_b_nk(Whh, 512, j0, kk + 64, lane);
        z0 = frag_b_nk(Whh, 512, j0 + 512, kk + 64, lane);
        g0 = frag_b_nk(Whh, 512, j0 + 1024, kk + 64, lane);
      }
      hr = WMMA_BF16(a1, r1, hr);
      hz = WMMA_BF16(a1, z1, hz);
      hn = WMMA_BF16(a1, g1, hn);
    }
    __syncthreads();  // all reads of h_b complete before overwrite

    if (t + 1 < T)  // warm L2/L0 for next step's xp rows
      __builtin_prefetch(xp + ((size_t)(lrow * T + t + 1)) * 1536 + j, 0, 1);

#pragma unroll
    for (int v = 0; v < 8; ++v) {
      const int b = v + 8 * ksel;  // batch row
      const float* x = xp + ((size_t)(b * T + t)) * 1536 + j;
      float r = sigmoidf_(x[0] + hr[v] + bhr);
      float z = sigmoidf_(x[512] + hz[v] + bhz);
      float n = tanhf(x[1024] + r * (hn[v] + bhn));
      float hold = h_f[b][j];
      float hnew = (1.f - z) * n + z * hold;
      h_f[b][j] = hnew;
      h_b[b][j] = (bf16_t)hnew;
      ys[((size_t)(b * T + t)) * ldy + j] = (bf16_t)hnew;
    }
    __syncthreads();
  }

  if (h_out)
    for (int i = tid; i < 16 * 512; i += 1024)
      h_out[i] = h_f[i >> 9][i & 511];
}

// ---------------------------------------------------------------------------
// Attention: one block per (b, 16 decoder steps). 256 thr / 8 waves.
//  scores[256,16] = en_out_b @ dec_tile^T (WMMA, K=512, double-buffered)
//  softmax over S in LDS, then ctx[16,512] = attn^T @ en_out_b (WMMA, K=256)
//  ctx written as bf16 into de[:, 512:1024].
// ---------------------------------------------------------------------------
__global__ __launch_bounds__(256) void k_attn(const bf16_t* __restrict__ enout,  // [16,256,512]
                                              bf16_t* __restrict__ de) {        // [16*256,1024]
  __shared__ float sc[256][17];
  const int b = blockIdx.y;
  const int t0 = blockIdx.x * 16;
  const int lane = threadIdx.x & 31, wave = threadIdx.x >> 5;
  const int lrow = lane & 15, ksel = lane >> 4;

  const bf16_t* E = enout + (size_t)b * 256 * 512;        // [256,512]
  const bf16_t* D = de + (size_t)(b * 256 + t0) * 1024;   // dec rows, ld=1024

  // Phase 1: scores (M = s, N = t). 16 M-tiles over 8 waves.
#pragma unroll
  for (int mi = 0; mi < 2; ++mi) {
    const int m0 = (wave * 2 + mi) * 16;
    v8f acc = {};
    const bf16_t* arow = E + (size_t)(m0 + lrow) * 512 + 8 * ksel;
    v16bf a0 = frag_a_rowmajor(arow);
    v16bf c0 = frag_b_nk(D, 1024, 0, 0, lane);
    for (int kk = 0; kk < 512; kk += 64) {
      v16bf a1 = frag_a_rowmajor(arow + kk + 32);
      v16bf c1 = frag_b_nk(D, 1024, 0, kk + 32, lane);
      acc = WMMA_BF16(a0, c0, acc);
      if (kk + 64 < 512) {
        a0 = frag_a_rowmajor(arow + kk + 64);
        c0 = frag_b_nk(D, 1024, 0, kk + 64, lane);
      }
      acc = WMMA_BF16(a1, c1, acc);
    }
#pragma unroll
    for (int v = 0; v < 8; ++v) sc[m0 + v + 8 * ksel][lrow] = acc[v];
  }
  __syncthreads();

  // Phase 2: softmax over S per column (16 columns, tiny).
  if (threadIdx.x < 16) {
    const int c = threadIdx.x;
    float mx = -3.0e38f;
    for (int s = 0; s < 256; ++s) mx = fmaxf(mx, sc[s][c]);
    float sum = 0.f;
    for (int s = 0; s < 256; ++s) { float e = __expf(sc[s][c] - mx); sc[s][c] = e; sum += e; }
    float inv = 1.f / sum;
    for (int s = 0; s < 256; ++s) sc[s][c] *= inv;
  }
  __syncthreads();

  // Phase 3: ctx (M = t, K = s, N = h). 32 N-tiles over 8 waves.
#pragma unroll
  for (int ni = 0; ni < 4; ++ni) {
    const int n0 = (wave * 4 + ni) * 16;
    v8f acc = {};
    for (int kk = 0; kk < 256; kk += 32) {
      v16bf a, bb;
#pragma unroll
      for (int i = 0; i < 16; ++i) {
        int ka = kk + i + 8 * ((i >= 8) + ksel);       // A[m=t, k=s] = attn[s][t]
        a[i] = (bf16_t)sc[ka][lrow];
        int kb = kk + i + 8 * ((i >= 8) + ksel);       // B[k=s, n=h] = en_out[s][h]
        bb[i] = E[(size_t)kb * 512 + n0 + lrow];
      }
      acc = WMMA_BF16(a, bb, acc);
    }
#pragma unroll
    for (int v = 0; v < 8; ++v)
      de[(size_t)(b * 256 + t0 + v + 8 * ksel) * 1024 + 512 + n0 + lrow] = (bf16_t)acc[v];
  }
}

// ---------------------------------------------------------------------------
// Host orchestration
// ---------------------------------------------------------------------------
extern "C" void kernel_launch(void* const* d_in, const int* in_sizes, int n_in,
                              void* d_out, int out_size, void* d_ws, size_t ws_size,
                              hipStream_t stream) {
  (void)in_sizes; (void)n_in; (void)out_size; (void)ws_size;
  constexpr int B = 16, S = 256, T = 256, E = 1024, H = 512, G = 3 * H; // 1536

  const int*   en_sen = (const int*)d_in[0];
  const int*   zh_sen = (const int*)d_in[1];
  const float* en_emb = (const float*)d_in[2];
  const float* zh_emb = (const float*)d_in[3];
  const float* h0     = (const float*)d_in[4];
  const float* Wih[4] = {(const float*)d_in[5],  (const float*)d_in[9],
                         (const float*)d_in[13], (const float*)d_in[17]};
  const float* Whh[4] = {(const float*)d_in[6],  (const float*)d_in[10],
                         (const float*)d_in[14], (const float*)d_in[18]};
  const float* bih[4] = {(const float*)d_in[7],  (const float*)d_in[11],
                         (const float*)d_in[15], (const float*)d_in[19]};
  const float* bhh[4] = {(const float*)d_in[8],  (const float*)d_in[12],
                         (const float*)d_in[16], (const float*)d_in[20]};
  const float* fcW = (const float*)d_in[21];
  const float* fcb = (const float*)d_in[22];
  float* out = (float*)d_out;

  // workspace carve-up (~150 MB total)
  char* w = (char*)d_ws;
  size_t off = 0;
  auto alloc = [&](size_t bytes) -> void* {
    void* p = w + off;
    off = (off + bytes + 255) & ~(size_t)255;
    return p;
  };
  bf16_t* exb   = (bf16_t*)alloc((size_t)B * S * E * 2);
  bf16_t* zxb   = (bf16_t*)alloc((size_t)B * T * E * 2);
  float*  xp    = (float*) alloc((size_t)B * S * G * 4);   // reused per layer
  bf16_t* y0    = (bf16_t*)alloc((size_t)B * S * H * 2);
  bf16_t* enout = (bf16_t*)alloc((size_t)B * S * H * 2);
  bf16_t* yd0   = (bf16_t*)alloc((size_t)B * T * H * 2);
  bf16_t* de    = (bf16_t*)alloc((size_t)B * T * E * 2);   // [dec | ctx]
  float*  hid   = (float*) alloc((size_t)2 * B * H * 4);
  const int Kin[4] = {E, H, E, H};
  bf16_t* Wihb[4]; bf16_t* Whhb[4];
  for (int l = 0; l < 4; ++l) {
    Wihb[l] = (bf16_t*)alloc((size_t)G * Kin[l] * 2);
    Whhb[l] = (bf16_t*)alloc((size_t)G * H * 2);
  }
  bf16_t* fcWb = (bf16_t*)alloc((size_t)32000 * E * 2);

  auto cast = [&](const float* s, bf16_t* d, size_t n) {
    k_cast_bf16<<<dim3((unsigned)((n + 255) / 256)), 256, 0, stream>>>(s, d, n);
  };
  for (int l = 0; l < 4; ++l) {
    cast(Wih[l], Wihb[l], (size_t)G * Kin[l]);
    cast(Whh[l], Whhb[l], (size_t)G * H);
  }
  cast(fcW, fcWb, (size_t)32000 * E);

  // embeddings
  {
    size_t n = (size_t)B * S * E;
    k_embed_enc<<<dim3((unsigned)((n + 255) / 256)), 256, 0, stream>>>(en_sen, en_emb, exb, B * S, E);
    k_embed_dec<<<dim3((unsigned)((n + 255) / 256)), 256, 0, stream>>>(zh_sen, zh_emb, zxb);
  }

  auto gemm = [&](const bf16_t* A, const bf16_t* Wt, const float* bias, float* C,
                  int M, int N, int K) {
    k_gemm_bf16<<<dim3(N / 128, M / 128), 256, 0, stream>>>(A, Wt, bias, C, M, N, K);
  };

  // encoder layer 0
  gemm(exb, Wihb[0], bih[0], xp, B * S, G, E);
  k_gru_scan<<<1, 1024, 0, stream>>>(xp, Whhb[0], bhh[0], h0, y0, H, hid, T);
  // encoder layer 1
  gemm(y0, Wihb[1], bih[1], xp, B * S, G, H);
  k_gru_scan<<<1, 1024, 0, stream>>>(xp, Whhb[1], bhh[1], h0 + (size_t)B * H, enout, H,
                                     hid + (size_t)B * H, T);
  // decoder layer 0 (teacher forcing)
  gemm(zxb, Wihb[2], bih[2], xp, B * T, G, E);
  k_gru_scan<<<1, 1024, 0, stream>>>(xp, Whhb[2], bhh[2], hid, yd0, H, nullptr, T);
  // decoder layer 1 -> cols 0..511 of de
  gemm(yd0, Wihb[3], bih[3], xp, B * T, G, H);
  k_gru_scan<<<1, 1024, 0, stream>>>(xp, Whhb[3], bhh[3], hid + (size_t)B * H, de, E, nullptr, T);

  // attention -> cols 512..1023 of de
  k_attn<<<dim3(T / 16, B), 256, 0, stream>>>(enout, de);

  // final vocab projection: [4096,1024] @ [32000,1024]^T + fcb -> d_out f32
  gemm(de, fcWb, fcb, out, B * T, 32000, E);
}
